// CEBlock_39127152066732
// MI455X (gfx1250) — matrix operand order repaired and checked by reading, not compiled, not run on hardware
//
#include <hip/hip_runtime.h>
#include <hip/hip_bf16.h>
#include <math.h>

// ---------------- problem constants ----------------
#define BB    32
#define LS    256
#define LT    128
#define CC    768
#define HH    12
#define DH    64
#define GRD   16
#define NTOK  384           // LT + LS
#define NKEEP 180           // ceil(0.7*256)
#define NREM  76
#define NNEW  308           // LT + NKEEP
#define C3    2304          // 3*C

typedef __attribute__((ext_vector_type(16))) _Float16 v16h;
typedef __attribute__((ext_vector_type(8)))  _Float16 v8h;
typedef __attribute__((ext_vector_type(8)))  float    v8f;

#define WMMA_F16(acc, a, b) \
  acc = __builtin_amdgcn_wmma_f32_16x16x32_f16(false, (a), false, (b), (short)0, (acc), false, false)

__device__ __forceinline__ float gelu_f(float x) {
  return 0.5f * x * (1.0f + erff(x * 0.70710678118654752f));
}
__device__ __forceinline__ int iabs_(int a) { return a < 0 ? -a : a; }

// ---- fragment loaders (CDNA5 wave32 16x16x32 f16 layouts) ----
// A 16x32 from f16 row: elems a[0..7]=k0+hi*8+{0..7}, a[8..15]=k0+16+hi*8+{0..7}
__device__ __forceinline__ v16h load_a16(const _Float16* __restrict__ Arow, int k0, int hi) {
  v8h lo = *(const v8h*)(Arow + k0 + (hi ? 8 : 0));
  v8h up = *(const v8h*)(Arow + k0 + 16 + (hi ? 8 : 0));
  v16h a;
#pragma unroll
  for (int t = 0; t < 8; ++t) { a[t] = lo[t]; a[8 + t] = up[t]; }
  return a;
}
// A 16x32 from f32 row (convert)
__device__ __forceinline__ v16h load_a32(const float* __restrict__ Arow, int k0, int hi) {
  v16h a;
#pragma unroll
  for (int v = 0; v < 8; ++v) {
    int k = k0 + 2 * (v & 3) + (hi ? 8 : 0) + ((v >= 4) ? 16 : 0);
    float2 p = *(const float2*)(Arow + k);
    a[2 * v] = (_Float16)p.x; a[2 * v + 1] = (_Float16)p.y;
  }
  return a;
}
// B 32x16 where B(k,n) is contiguous along k (transposed-weight f16 row): b[0..7]=k0+hi*8+{0..7}, b[8..15]=k0+16+hi*8+{0..7}
__device__ __forceinline__ v16h load_bt16(const _Float16* __restrict__ Brow, int k0, int hi) {
  v8h lo = *(const v8h*)(Brow + k0 + (hi ? 8 : 0));
  v8h up = *(const v8h*)(Brow + k0 + 16 + (hi ? 8 : 0));
  v16h b;
#pragma unroll
  for (int t = 0; t < 8; ++t) { b[t] = lo[t]; b[8 + t] = up[t]; }
  return b;
}

// ================= weight convert + transpose:  wT[n*K+k] = (f16) w[k*N+n] =================
__global__ void wcvt_kernel(const float* __restrict__ w, _Float16* __restrict__ wT, int K, int N) {
  int k = blockIdx.x * 16 + (threadIdx.x & 15);
  int n = blockIdx.y * 16 + (threadIdx.x >> 4);
  wT[(size_t)n * K + k] = (_Float16)w[(size_t)k * N + n];
}

// ================= max-pooled template reps =================
__global__ void rep_kernel(const float* __restrict__ x, const float* __restrict__ ltm,
                           float* __restrict__ rep0, float* __restrict__ rep1) {
  int b = blockIdx.x;
  int c = blockIdx.y * 256 + threadIdx.x;          // C = 768 = 3*256
  float m0 = -INFINITY, m1 = -INFINITY;
  for (int i = 0; i < LT / 2; ++i) {
    m0 = fmaxf(m0, x[((size_t)(b * NTOK + i)) * CC + c] * ltm[b * 64 + i]);
    m1 = fmaxf(m1, x[((size_t)(b * NTOK + 64 + i)) * CC + c] * ltm[BB * 64 + b * 64 + i]);
  }
  rep0[b * CC + c] = m0;
  rep1[b * CC + c] = m1;
}

// ========== d1: feat(2304)->768, gelu; 3 branch-free K segments, 2x2 WMMA blocking ==========
__global__ void d1_kernel(const float* __restrict__ x, const float* __restrict__ rep0,
                          const float* __restrict__ rep1, const _Float16* __restrict__ wT,
                          const float* __restrict__ bias, _Float16* __restrict__ out) {
  int lane = threadIdx.x & 31;
  int n0 = blockIdx.x << 5, m0 = blockIdx.y << 5;   // 32x32 tile
  int m = lane & 15, n = lane & 15, hi = (lane >> 4) & 1;
  int mg = m0 + m;
  int b = mg >> 8, i = mg & 255;                    // 32-row tile stays inside one batch
  const float* xr0 = x + ((size_t)(b * NTOK + LT + i)) * CC;
  const float* xr1 = xr0 + (size_t)16 * CC;
  const float* r0 = rep0 + b * CC;
  const float* r1 = rep1 + b * CC;
  const _Float16* Br0 = wT + (size_t)(n0 + n) * C3;       // row stride K=2304
  const _Float16* Br1 = Br0 + (size_t)16 * C3;
  v8f a00 = {0,0,0,0,0,0,0,0}, a01 = {0,0,0,0,0,0,0,0};
  v8f a10 = {0,0,0,0,0,0,0,0}, a11 = {0,0,0,0,0,0,0,0};
  // segment 0: k in [0,768) from x
  for (int k0 = 0; k0 < CC; k0 += 32) {
    v16h fa0 = load_a32(xr0, k0, hi);
    v16h fa1 = load_a32(xr1, k0, hi);
    v16h fb0 = load_bt16(Br0, k0, hi);
    v16h fb1 = load_bt16(Br1, k0, hi);
    WMMA_F16(a00, fa0, fb0); WMMA_F16(a01, fa0, fb1);
    WMMA_F16(a10, fa1, fb0); WMMA_F16(a11, fa1, fb1);
  }
  // segment 1: k in [768,1536) from rep0 (row-independent)
  for (int k0 = 0; k0 < CC; k0 += 32) {
    v16h fa  = load_a32(r0, k0, hi);
    v16h fb0 = load_bt16(Br0 + CC, k0, hi);
    v16h fb1 = load_bt16(Br1 + CC, k0, hi);
    WMMA_F16(a00, fa, fb0); WMMA_F16(a01, fa, fb1);
    WMMA_F16(a10, fa, fb0); WMMA_F16(a11, fa, fb1);
  }
  // segment 2: k in [1536,2304) from rep1
  for (int k0 = 0; k0 < CC; k0 += 32) {
    v16h fa  = load_a32(r1, k0, hi);
    v16h fb0 = load_bt16(Br0 + 2 * CC, k0, hi);
    v16h fb1 = load_bt16(Br1 + 2 * CC, k0, hi);
    WMMA_F16(a00, fa, fb0); WMMA_F16(a01, fa, fb1);
    WMMA_F16(a10, fa, fb0); WMMA_F16(a11, fa, fb1);
  }
  float b0 = bias[n0 + n], b1 = bias[n0 + 16 + n];
#pragma unroll
  for (int r = 0; r < 8; ++r) {
    int mm = m0 + r + (hi ? 8 : 0);
    out[(size_t)mm * CC + n0 + n]             = (_Float16)gelu_f(a00[r] + b0);
    out[(size_t)mm * CC + n0 + 16 + n]        = (_Float16)gelu_f(a01[r] + b1);
    out[(size_t)(mm + 16) * CC + n0 + n]      = (_Float16)gelu_f(a10[r] + b0);
    out[(size_t)(mm + 16) * CC + n0 + 16 + n] = (_Float16)gelu_f(a11[r] + b1);
  }
}

// ========== generic f16-in WMMA GEMM, 2x2 blocking, transposed f16 weights ==========
// requires M%32==0, N%32==0, K%32==0; A row-major f16 (lda halves), Bt[n*K+k] f16
// v = (act? gelu:id)( A@B + bias[n] ) + res[m*ldo+n] ; stored to outh (f16) or outf (f32)
__global__ void gemm_kernel(const _Float16* __restrict__ A, const _Float16* __restrict__ Bt,
                            const float* __restrict__ bias, const float* __restrict__ res,
                            _Float16* __restrict__ outh, float* __restrict__ outf,
                            int K, int lda, int ldo, int act) {
  int lane = threadIdx.x & 31;
  int n0 = blockIdx.x << 5, m0 = blockIdx.y << 5;
  int m = lane & 15, n = lane & 15, hi = (lane >> 4) & 1;
  const _Float16* Ar0 = A + (size_t)(m0 + m) * lda;
  const _Float16* Ar1 = Ar0 + (size_t)16 * lda;
  const _Float16* Br0 = Bt + (size_t)(n0 + n) * K;
  const _Float16* Br1 = Br0 + (size_t)16 * K;
  v8f a00 = {0,0,0,0,0,0,0,0}, a01 = {0,0,0,0,0,0,0,0};
  v8f a10 = {0,0,0,0,0,0,0,0}, a11 = {0,0,0,0,0,0,0,0};
  for (int k0 = 0; k0 < K; k0 += 32) {
    v16h fa0 = load_a16(Ar0, k0, hi);
    v16h fa1 = load_a16(Ar1, k0, hi);
    v16h fb0 = load_bt16(Br0, k0, hi);
    v16h fb1 = load_bt16(Br1, k0, hi);
    WMMA_F16(a00, fa0, fb0); WMMA_F16(a01, fa0, fb1);
    WMMA_F16(a10, fa1, fb0); WMMA_F16(a11, fa1, fb1);
  }
  float b0 = bias ? bias[n0 + n] : 0.f;
  float b1 = bias ? bias[n0 + 16 + n] : 0.f;
#pragma unroll
  for (int r = 0; r < 8; ++r) {
    int mm = m0 + r + (hi ? 8 : 0);
    float v00 = a00[r] + b0, v01 = a01[r] + b1;
    float v10 = a10[r] + b0, v11 = a11[r] + b1;
    if (act == 1) { v00 = gelu_f(v00); v01 = gelu_f(v01); v10 = gelu_f(v10); v11 = gelu_f(v11); }
    if (res) {
      v00 += res[(size_t)mm * ldo + n0 + n];
      v01 += res[(size_t)mm * ldo + n0 + 16 + n];
      v10 += res[(size_t)(mm + 16) * ldo + n0 + n];
      v11 += res[(size_t)(mm + 16) * ldo + n0 + 16 + n];
    }
    if (outh) {
      outh[(size_t)mm * ldo + n0 + n]             = (_Float16)v00;
      outh[(size_t)mm * ldo + n0 + 16 + n]        = (_Float16)v01;
      outh[(size_t)(mm + 16) * ldo + n0 + n]      = (_Float16)v10;
      outh[(size_t)(mm + 16) * ldo + n0 + 16 + n] = (_Float16)v11;
    } else {
      outf[(size_t)mm * ldo + n0 + n]             = v00;
      outf[(size_t)mm * ldo + n0 + 16 + n]        = v01;
      outf[(size_t)(mm + 16) * ldo + n0 + n]      = v10;
      outf[(size_t)(mm + 16) * ldo + n0 + 16 + n] = v11;
    }
  }
}

// ================= d4 head + sigmoid =================
__global__ void d4_kernel(const _Float16* __restrict__ h3, const float* __restrict__ w,
                          const float* __restrict__ bias, float* __restrict__ probs) {
  int r = blockIdx.x * 256 + threadIdx.x;
  if (r >= BB * LS) return;
  float acc = bias[0];
  for (int k = 0; k < 192; ++k) acc += (float)h3[(size_t)r * 192 + k] * w[k];
  probs[r] = 1.0f / (1.0f + __expf(-acc));
}

// ========== conv3x3 + argmax + windows + stable top-k + policy ==========
__global__ void select_kernel(const float* __restrict__ probs, const int* __restrict__ gis,
                              float* __restrict__ dkeep, float* __restrict__ drem,
                              int* __restrict__ order, int* __restrict__ keybg) {
  int b = blockIdx.x, tid = threadIdx.x;
  __shared__ float sp[256];
  __shared__ float skey[256];
  __shared__ int   sidx[256];
  __shared__ int   scx, scy;
  sp[tid] = probs[b * 256 + tid];
  __syncthreads();
  int gx = tid >> 4, gy = tid & 15;
  float f = 0.f;
  for (int dx = -1; dx <= 1; ++dx)
    for (int dy = -1; dy <= 1; ++dy) {
      int xx = gx + dx, yy = gy + dy;
      if (xx >= 0 && xx < GRD && yy >= 0 && yy < GRD) f += sp[xx * GRD + yy];
    }
  skey[tid] = f; sidx[tid] = tid;
  __syncthreads();
  for (int s = 128; s > 0; s >>= 1) {            // argmax, first index on tie
    if (tid < s) {
      if (skey[tid + s] > skey[tid] ||
          (skey[tid + s] == skey[tid] && sidx[tid + s] < sidx[tid])) {
        skey[tid] = skey[tid + s]; sidx[tid] = sidx[tid + s];
      }
    }
    __syncthreads();
  }
  if (tid == 0) { scx = sidx[0] >> 4; scy = sidx[0] & 15; }
  __syncthreads();
  int cx = scx, cy = scy;
  bool m11 = (iabs_(gx - cx) <= 5) && (iabs_(gy - cy) <= 5);
  skey[tid] = m11 ? -INFINITY : -sp[tid];        // sort key = -p_out (ascending)
  sidx[tid] = tid;
  __syncthreads();
  // bitonic sort by (key, idx) ascending == stable argsort(-p_out)
  for (int k = 2; k <= 256; k <<= 1)
    for (int j = k >> 1; j > 0; j >>= 1) {
      int ixj = tid ^ j;
      if (ixj > tid) {
        bool up = ((tid & k) == 0);
        float k1 = skey[tid], k2 = skey[ixj];
        int i1 = sidx[tid], i2 = sidx[ixj];
        bool sw = up ? (k1 > k2 || (k1 == k2 && i1 > i2))
                     : (k1 < k2 || (k1 == k2 && i1 < i2));
        if (sw) { skey[tid] = k2; skey[ixj] = k1; sidx[tid] = i2; sidx[ixj] = i1; }
      }
      __syncthreads();
    }
  order[b * 256 + tid] = sidx[tid];
  if (tid < NKEEP) {
    int jj = sidx[tid];
    dkeep[b * NKEEP + tid] = (float)gis[b * 256 + jj];
    int gxj = jj >> 4, gyj = jj & 15;
    int m7 = (iabs_(gxj - cx) <= 3) && (iabs_(gyj - cy) <= 3);
    int sf = (sp[jj] > 0.5f && m7) ? 1 : 0;
    keybg[b * NNEW + LT + tid] = (sf == 0) ? 1 : 0;   // P[:,:,2] > 0.5
  } else {
    drem[b * NREM + (tid - NKEEP)] = (float)gis[b * 256 + sidx[tid]];
  }
  if (tid < LT) keybg[b * NNEW + tid] = 0;            // template/dynamic rows
}

// ================= gather pruned tokens -> xnew (f32) =================
__global__ void gather_kernel(const float* __restrict__ x, const int* __restrict__ order,
                              float* __restrict__ xnew) {
  int b = blockIdx.x, r = blockIdx.y, tid = threadIdx.x;
  int src = (r < LT) ? r : (LT + order[b * 256 + (r - LT)]);
  const float* ps = x + ((size_t)(b * NTOK + src)) * CC;
  float* pd = xnew + ((size_t)(b * NNEW + r)) * CC;
  for (int c = tid; c < CC; c += 256) pd[c] = ps[c];
}

// ================= LayerNorm (row per block), f32 in -> f16 out =================
__global__ void ln_kernel(const float* __restrict__ in, _Float16* __restrict__ out,
                          const float* __restrict__ w, const float* __restrict__ bias) {
  int row = blockIdx.x, tid = threadIdx.x;
  const float* p = in + (size_t)row * CC;
  __shared__ float s1[256], s2[256];
  float a = 0.f, b2 = 0.f;
  for (int c = tid; c < CC; c += 256) { float v = p[c]; a += v; b2 += v * v; }
  s1[tid] = a; s2[tid] = b2;
  __syncthreads();
  for (int s = 128; s > 0; s >>= 1) {
    if (tid < s) { s1[tid] += s1[tid + s]; s2[tid] += s2[tid + s]; }
    __syncthreads();
  }
  float mu = s1[0] / (float)CC;
  float var = s2[0] / (float)CC - mu * mu;
  float inv = rsqrtf(var + 1e-5f);
  for (int c = tid; c < CC; c += 256)
    out[(size_t)row * CC + c] = (_Float16)((p[c] - mu) * inv * w[c] + bias[c]);
}

// ========== scores = (Q K^T)/8 with key_bg mask; f16 QKV, WMMA ==========
__global__ void scores_kernel(const _Float16* __restrict__ qkv, const int* __restrict__ keybg,
                              float* __restrict__ attn) {
  int lane = threadIdx.x & 31;
  int n0 = blockIdx.x << 4, m0 = blockIdx.y << 4;
  int z = blockIdx.z, b = z / HH, h = z % HH;
  const _Float16* Q  = qkv + (size_t)b * NNEW * C3 + (size_t)h * DH;       // Q[m*C3+k]
  const _Float16* Kb = qkv + (size_t)b * NNEW * C3 + CC + (size_t)h * DH;  // K[n*C3+k]
  int m = lane & 15, n = lane & 15, hi = (lane >> 4) & 1;
  int mrow = m0 + m; if (mrow > NNEW - 1) mrow = NNEW - 1;   // clamp -> in-bounds
  int nrow = n0 + n; if (nrow > NNEW - 1) nrow = NNEW - 1;
  const _Float16* Qr = Q + (size_t)mrow * C3;
  const _Float16* Kr = Kb + (size_t)nrow * C3;
  v8f acc = {0,0,0,0,0,0,0,0};
#pragma unroll
  for (int k0 = 0; k0 < DH; k0 += 32) {
    v16h fa = load_a16(Qr, k0, hi);
    v16h fb = load_bt16(Kr, k0, hi);
    WMMA_F16(acc, fa, fb);
  }
  int nn = n0 + n;
  if (nn < NNEW) {
    bool bg = keybg[b * NNEW + nn] != 0;
#pragma unroll
    for (int r = 0; r < 8; ++r) {
      int mm = m0 + r + (hi ? 8 : 0);
      if (mm < NNEW) {
        float s = acc[r] * 0.125f;                  // dh^-0.5
        if (bg) s = -1.0e9f;
        attn[((size_t)(b * HH + h) * NNEW + mm) * NNEW + nn] = s;
      }
    }
  }
}

// ================= row softmax in place (f32) =================
__global__ void softmax_kernel(float* __restrict__ attn) {
  int row = blockIdx.x, tid = threadIdx.x;
  float* p = attn + (size_t)row * NNEW;
  __shared__ float sr[128];
  float mx = -INFINITY;
  for (int j = tid; j < NNEW; j += 128) mx = fmaxf(mx, p[j]);
  sr[tid] = mx;
  __syncthreads();
  for (int s = 64; s > 0; s >>= 1) { if (tid < s) sr[tid] = fmaxf(sr[tid], sr[tid + s]); __syncthreads(); }
  mx = sr[0];
  __syncthreads();
  float sum = 0.f;
  for (int j = tid; j < NNEW; j += 128) { float e = __expf(p[j] - mx); p[j] = e; sum += e; }
  sr[tid] = sum;
  __syncthreads();
  for (int s = 64; s > 0; s >>= 1) { if (tid < s) sr[tid] += sr[tid + s]; __syncthreads(); }
  float inv = 1.0f / sr[0];
  for (int j = tid; j < NNEW; j += 128) p[j] *= inv;
}

// ========== out = attn @ V (K=308: 9 full steps + guarded tail), WMMA, f16 out ==========
__global__ void attnv_kernel(const float* __restrict__ attn, const _Float16* __restrict__ qkv,
                             _Float16* __restrict__ outb) {
  int lane = threadIdx.x & 31;
  int n0 = blockIdx.x << 4, m0 = blockIdx.y << 4;
  int z = blockIdx.z, b = z / HH, h = z % HH;
  const float* A = attn + (size_t)(b * HH + h) * NNEW * NNEW;                 // A[m*308+k]
  const _Float16* V = qkv + (size_t)b * NNEW * C3 + 2 * CC + (size_t)h * DH;  // V[k*C3+n]
  int m = lane & 15, n = lane & 15, hi = (lane >> 4) & 1;
  int mrow = m0 + m; if (mrow > NNEW - 1) mrow = NNEW - 1;
  const float* Arow = A + (size_t)mrow * NNEW;
  int nn = n0 + n;                                      // nn < 64, always valid
  const _Float16* Vc = V + nn;
  v8f acc = {0,0,0,0,0,0,0,0};
  for (int k0 = 0; k0 < 288; k0 += 32) {                // 9 unguarded steps
    v16h fa = load_a32(Arow, k0, hi);
    v16h fb;
#pragma unroll
    for (int v = 0; v < 8; ++v) {
      int k = k0 + 2 * (v & 3) + (hi ? 8 : 0) + ((v >= 4) ? 16 : 0);
      fb[2 * v]     = Vc[(size_t)k * C3];
      fb[2 * v + 1] = Vc[(size_t)(k + 1) * C3];
    }
    WMMA_F16(acc, fa, fb);
  }
  {                                                     // tail: k in [288,320), valid < 308
    const int k0 = 288;
    v16h fa, fb;
#pragma unroll
    for (int v = 0; v < 8; ++v) {
      int k = k0 + 2 * (v & 3) + (hi ? 8 : 0) + ((v >= 4) ? 16 : 0);
      fa[2 * v]     = (_Float16)((k < NNEW) ? Arow[k] : 0.f);
      fa[2 * v + 1] = (_Float16)((k + 1 < NNEW) ? Arow[k + 1] : 0.f);
    }
#pragma unroll
    for (int v = 0; v < 8; ++v) {
      int k = k0 + 2 * (v & 3) + (hi ? 8 : 0) + ((v >= 4) ? 16 : 0);
      fb[2 * v]     = (k < NNEW) ? Vc[(size_t)k * C3] : (_Float16)0.f;
      fb[2 * v + 1] = (k + 1 < NNEW) ? Vc[(size_t)(k + 1) * C3] : (_Float16)0.f;
    }
    WMMA_F16(acc, fa, fb);
  }
#pragma unroll
  for (int r = 0; r < 8; ++r) {
    int mm = m0 + r + (hi ? 8 : 0);
    if (mm < NNEW)
      outb[(size_t)(b * NNEW + mm) * CC + h * DH + nn] = (_Float16)acc[r];
  }
}

// ===================================================================
extern "C" void kernel_launch(void* const* d_in, const int* in_sizes, int n_in,
                              void* d_out, int out_size, void* d_ws, size_t ws_size,
                              hipStream_t stream) {
  const float* x      = (const float*)d_in[0];
  const int*   gis    = (const int*)d_in[2];
  const float* ltm    = (const float*)d_in[3];
  const float* ln1_w  = (const float*)d_in[4];
  const float* ln1_b  = (const float*)d_in[5];
  const float* qkv_w  = (const float*)d_in[6];
  const float* proj_w = (const float*)d_in[7];
  const float* proj_b = (const float*)d_in[8];
  const float* ln2_w  = (const float*)d_in[9];
  const float* ln2_b  = (const float*)d_in[10];
  const float* fc1_w  = (const float*)d_in[11];
  const float* fc1_b  = (const float*)d_in[12];
  const float* fc2_w  = (const float*)d_in[13];
  const float* fc2_b  = (const float*)d_in[14];
  const float* d1_w   = (const float*)d_in[15];
  const float* d1_b   = (const float*)d_in[16];
  const float* d2_w   = (const float*)d_in[17];
  const float* d2_b   = (const float*)d_in[18];
  const float* d3_w   = (const float*)d_in[19];
  const float* d3_b   = (const float*)d_in[20];
  const float* d4_w   = (const float*)d_in[21];
  const float* d4_b   = (const float*)d_in[22];

  // ---- d_out layout (floats): xout | keep | removed | attn ----
  float* outf  = (float*)d_out;
  float* xout  = outf;                                  // B*308*768
  float* dkeep = outf + (size_t)BB * NNEW * CC;         // B*180
  float* drem  = dkeep + (size_t)BB * NKEEP;            // B*76
  float* attn  = drem + (size_t)BB * NREM;              // B*12*308*308

  // ---- workspace: bump allocator, 256B aligned ----
  size_t off = 0;
  char* base = (char*)d_ws;
  auto alloc = [&](size_t bytes) -> void* {
    void* p = base + off;
    off = (off + bytes + 255) & ~(size_t)255;
    return p;
  };
  float* rep0  = (float*)alloc((size_t)BB * CC * 4);
  float* rep1  = (float*)alloc((size_t)BB * CC * 4);
  float* probs = (float*)alloc((size_t)BB * LS * 4);
  int*   order = (int*)alloc((size_t)BB * LS * 4);
  int*   keybg = (int*)alloc((size_t)BB * NNEW * 4);
  float* xnew  = (float*)alloc((size_t)BB * NNEW * CC * 4);     // residual stream (f32)
  _Float16* h1h   = (_Float16*)alloc((size_t)BB * LS * CC * 2);
  _Float16* h2h   = (_Float16*)alloc((size_t)BB * LS * 384 * 2);
  _Float16* h3h   = (_Float16*)alloc((size_t)BB * LS * 192 * 2);
  _Float16* hnh   = (_Float16*)alloc((size_t)BB * NNEW * CC * 2);   // ln1 out; reused as attn-out
  _Float16* qkvh  = (_Float16*)alloc((size_t)BB * NNEW * C3 * 2);   // reused as ln2 out
  _Float16* mlp1h = (_Float16*)alloc((size_t)BB * NNEW * 3072 * 2);
  _Float16* d1t   = (_Float16*)alloc((size_t)C3 * CC * 2);
  _Float16* d2t   = (_Float16*)alloc((size_t)CC * 384 * 2);
  _Float16* d3t   = (_Float16*)alloc((size_t)384 * 192 * 2);
  _Float16* qkvt  = (_Float16*)alloc((size_t)CC * C3 * 2);
  _Float16* projt = (_Float16*)alloc((size_t)CC * CC * 2);
  _Float16* fc1t  = (_Float16*)alloc((size_t)CC * 3072 * 2);
  _Float16* fc2t  = (_Float16*)alloc((size_t)3072 * CC * 2);
  _Float16* attnoh = hnh;                               // reuse after qkv GEMM
  _Float16* mbufh  = qkvh;                              // reuse after attn@V

  // 0) weight f16 transposed copies (reused ~300x each)
  wcvt_kernel<<<dim3(C3 / 16, CC / 16), 256, 0, stream>>>(d1_w, d1t, C3, CC);
  wcvt_kernel<<<dim3(CC / 16, 384 / 16), 256, 0, stream>>>(d2_w, d2t, CC, 384);
  wcvt_kernel<<<dim3(384 / 16, 192 / 16), 256, 0, stream>>>(d3_w, d3t, 384, 192);
  wcvt_kernel<<<dim3(CC / 16, C3 / 16), 256, 0, stream>>>(qkv_w, qkvt, CC, C3);
  wcvt_kernel<<<dim3(CC / 16, CC / 16), 256, 0, stream>>>(proj_w, projt, CC, CC);
  wcvt_kernel<<<dim3(CC / 16, 3072 / 16), 256, 0, stream>>>(fc1_w, fc1t, CC, 3072);
  wcvt_kernel<<<dim3(3072 / 16, CC / 16), 256, 0, stream>>>(fc2_w, fc2t, 3072, CC);

  // 1) masked max-pooled reps
  rep_kernel<<<dim3(BB, 3), 256, 0, stream>>>(x, ltm, rep0, rep1);
  // 2) score MLP
  d1_kernel<<<dim3(CC / 32, (BB * LS) / 32), 32, 0, stream>>>(x, rep0, rep1, d1t, d1_b, h1h);
  gemm_kernel<<<dim3(384 / 32, (BB * LS) / 32), 32, 0, stream>>>(
      h1h, d2t, d2_b, nullptr, h2h, nullptr, CC, CC, 384, 1);
  gemm_kernel<<<dim3(192 / 32, (BB * LS) / 32), 32, 0, stream>>>(
      h2h, d3t, d3_b, nullptr, h3h, nullptr, 384, 384, 192, 1);
  d4_kernel<<<(BB * LS) / 256, 256, 0, stream>>>(h3h, d4_w, d4_b, probs);
  // 3) conv + argmax + windows + stable top-k
  select_kernel<<<BB, 256, 0, stream>>>(probs, gis, dkeep, drem, order, keybg);
  // 4) pruned token gather
  gather_kernel<<<dim3(BB, NNEW), 256, 0, stream>>>(x, order, xnew);
  // 5) attention block
  ln_kernel<<<BB * NNEW, 256, 0, stream>>>(xnew, hnh, ln1_w, ln1_b);
  gemm_kernel<<<dim3(C3 / 32, (BB * NNEW) / 32), 32, 0, stream>>>(
      hnh, qkvt, nullptr, nullptr, qkvh, nullptr, CC, CC, C3, 0);
  scores_kernel<<<dim3(20, 20, BB * HH), 32, 0, stream>>>(qkvh, keybg, attn);
  softmax_kernel<<<BB * HH * NNEW, 128, 0, stream>>>(attn);
  attnv_kernel<<<dim3(DH / 16, 20, BB * HH), 32, 0, stream>>>(attn, qkvh, attnoh);
  gemm_kernel<<<dim3(CC / 32, (BB * NNEW) / 32), 32, 0, stream>>>(
      attnoh, projt, proj_b, xnew, nullptr, xnew, CC, CC, CC, 0);
  // 6) MLP block
  ln_kernel<<<BB * NNEW, 256, 0, stream>>>(xnew, mbufh, ln2_w, ln2_b);
  gemm_kernel<<<dim3(3072 / 32, (BB * NNEW) / 32), 32, 0, stream>>>(
      mbufh, fc1t, fc1_b, nullptr, mlp1h, nullptr, CC, CC, 3072, 1);
  gemm_kernel<<<dim3(CC / 32, (BB * NNEW) / 32), 32, 0, stream>>>(
      mlp1h, fc2t, fc2_b, xnew, nullptr, xout, 3072, 3072, CC, 0);
}
